// SAGELayer_52304111730952
// MI455X (gfx1250) — compile-verified, matrix operand order checked by reading, not run on hardware
//
#include <hip/hip_runtime.h>
#include <hip/hip_bf16.h>
#include <stdint.h>

#define NN 50000
#define NE 800000
#define DD 128
#define KK 256  /* 2*D */

typedef __attribute__((ext_vector_type(16))) __bf16 v16bf;
typedef __attribute__((ext_vector_type(8)))  float  v8f;

// ---- workspace layout (bytes) ----
// 0         : msg_sum  (NN*DD f32 = 25,600,000)
// 25600000  : deg      (NN f32   =    200,000)
// 25800000  : wmsg_hi  (32768 bf16 = 65536)
// 25865536  : wmsg_lo
// 25931072  : wapp_hi
// 25996608  : wapp_lo   (total 26,062,144 B)

// ---- CDNA5 async global->LDS copy (ASYNCcnt-tracked) ----
// Generic pointers to __shared__ carry the LDS allocation offset in addr[31:0]
// (ISA 10.2 aperture rule), which is exactly what VDST of the async op needs.
__device__ __forceinline__ void async_copy_b128(void* lds_ptr, const void* gptr) {
    uint32_t lds_off = (uint32_t)(uintptr_t)lds_ptr;
    asm volatile("global_load_async_to_lds_b128 %0, %1, off"
                 :: "v"(lds_off), "v"((uint64_t)(uintptr_t)gptr)
                 : "memory");
}
__device__ __forceinline__ void wait_async0() {
    asm volatile("s_wait_asynccnt 0" ::: "memory");
}

__global__ void sage_zero(float* __restrict__ p, int n) {
    int i = blockIdx.x * blockDim.x + threadIdx.x;
    if (i < n) p[i] = 0.0f;
}

// Pack W (row-major [DD, KK], msg = x @ W^T) into B-fragment-major bf16 hi/lo.
// Fragment (kt,nt): 32 lanes x 16 bf16. lane l -> n = nt*16+(l&15), kbase = kt*32+(l>>4)*16.
__global__ void sage_wfrag(const float* __restrict__ Wmsg, const float* __restrict__ Wapp,
                           __bf16* __restrict__ mh, __bf16* __restrict__ ml,
                           __bf16* __restrict__ ah, __bf16* __restrict__ al) {
    int t = blockIdx.x * blockDim.x + threadIdx.x;  // 65536 total
    if (t >= 65536) return;
    int w    = t >> 15;
    int r    = t & 32767;
    int i    = r & 15;
    int lane = (r >> 4) & 31;
    int nt   = (r >> 9) & 7;
    int kt   = (r >> 12) & 7;
    int n = nt * 16 + (lane & 15);
    int k = kt * 32 + ((lane >> 4) << 4) + i;
    const float* W = w ? Wapp : Wmsg;
    float f  = W[n * KK + k];
    __bf16 h = (__bf16)f;
    __bf16 l = (__bf16)(f - (float)h);
    if (w) { ah[r] = h; al[r] = l; } else { mh[r] = h; ml[r] = l; }
}

__global__ void sage_deg(const int* __restrict__ dst, float* __restrict__ deg) {
    int i = blockIdx.x * blockDim.x + threadIdx.x;
    if (i < NE) atomicAdd(&deg[dst[i]], 1.0f);
}

// hi/lo bf16 split with optional pre-scale (bf16x3 ~= fp32 accuracy)
#define CVT4S(F, B, S)                                                      \
    { __bf16 h_; float x_;                                                  \
      x_ = (F).x * (S); h_ = (__bf16)x_; ahv[(B)+0] = h_; alv[(B)+0] = (__bf16)(x_ - (float)h_); \
      x_ = (F).y * (S); h_ = (__bf16)x_; ahv[(B)+1] = h_; alv[(B)+1] = (__bf16)(x_ - (float)h_); \
      x_ = (F).z * (S); h_ = (__bf16)x_; ahv[(B)+2] = h_; alv[(B)+2] = (__bf16)(x_ - (float)h_); \
      x_ = (F).w * (S); h_ = (__bf16)x_; ahv[(B)+3] = h_; alv[(B)+3] = (__bf16)(x_ - (float)h_); }

#define WMMA_BF16(A, B, C) \
    __builtin_amdgcn_wmma_f32_16x16x32_bf16(false, (A), false, (B), (short)0, (C), false, false)

// Edge message GEMM fused with segment-sum scatter.
// Block = 256 threads (8 waves). Each block: 16 tiles of 16 edges, double-buffered
// LDS staging via async global->LDS copies overlapped with WMMA compute.
__global__ __launch_bounds__(256) void sage_edge(
    const float* __restrict__ nfeats, const float* __restrict__ efeats,
    const int* __restrict__ src, const int* __restrict__ dst,
    const v16bf* __restrict__ bfh, const v16bf* __restrict__ bfl,
    const float* __restrict__ bias, float* __restrict__ msg_sum) {
    __shared__ float At[2][16 * 260];  // 2 x (16 rows x 256 f32, stride 260)
    const int tid   = threadIdx.x;
    const int lane  = tid & 31;
    const int wv    = tid >> 5;      // n-tile id 0..7
    const int m     = lane & 15;
    const int hh    = lane >> 4;     // 0 or 1
    const int ncol  = wv * 16 + m;
    const float bv  = bias[ncol];
    const float4* nf4 = (const float4*)nfeats;
    const float4* ef4 = (const float4*)efeats;
    const int tile0 = blockIdx.x * 16;

    // issue async staging of tile (tile0+t) into buffer buf
    auto stage = [&](int t, int buf) {
        const int base_e = (tile0 + t) * 16;
        #pragma unroll
        for (int s = 0; s < 4; ++s) {
            int j   = tid + s * 256;   // 1024 float4 per tile
            int row = j >> 6;
            int c4  = j & 63;
            const float4* gp = (c4 < 32)
                ? &nf4[(size_t)src[base_e + row] * 32 + c4]
                : &ef4[(size_t)(base_e + row) * 32 + (c4 - 32)];
            async_copy_b128(&At[buf][row * 260 + c4 * 4], gp);
        }
    };

    stage(0, 0);
    for (int t = 0; t < 16; ++t) {
        const int cur = t & 1;
        wait_async0();      // this wave's async writes (buffer `cur`) complete
        __syncthreads();    // all waves' staging complete; prior compute done
        if (t < 15) stage(t + 1, cur ^ 1);  // uniform branch: prefetch next tile

        v8f acc;
        #pragma unroll
        for (int q = 0; q < 8; ++q) acc[q] = bv;

        #pragma unroll
        for (int kt = 0; kt < 8; ++kt) {
            const float* ap = &At[cur][m * 260 + kt * 32 + hh * 16];
            float4 f0 = *(const float4*)(ap + 0);
            float4 f1 = *(const float4*)(ap + 4);
            float4 f2 = *(const float4*)(ap + 8);
            float4 f3 = *(const float4*)(ap + 12);
            v16bf ahv, alv;
            CVT4S(f0, 0, 1.0f) CVT4S(f1, 4, 1.0f) CVT4S(f2, 8, 1.0f) CVT4S(f3, 12, 1.0f)
            const int fi = (kt * 8 + wv) * 32 + lane;
            v16bf bh = bfh[fi];
            v16bf bl = bfl[fi];
            acc = WMMA_BF16(ahv, bh, acc);
            acc = WMMA_BF16(alv, bh, acc);
            acc = WMMA_BF16(ahv, bl, acc);
        }
        // Scatter accumulator: VGPR q, lanes 0-15 -> M=q ; lanes 16-31 -> M=q+8
        const int base_e = (tile0 + t) * 16;
        const int roff = hh * 8;
        #pragma unroll
        for (int q = 0; q < 8; ++q) {
            int e = base_e + q + roff;
            atomicAdd(&msg_sum[(size_t)dst[e] * DD + ncol], acc[q]);
        }
    }
}

// Apply GEMM: relu([nfeats || msg_sum/deg] @ Wapp^T + b). One 16-node tile / block.
// Staging is a raw async copy; the 1/deg mean-scale is applied per-row during
// fragment conversion (kt>=4 <=> the h_neigh half of K).
__global__ __launch_bounds__(256) void sage_apply(
    const float* __restrict__ nfeats, const float* __restrict__ msg_sum,
    const float* __restrict__ deg,
    const v16bf* __restrict__ bfh, const v16bf* __restrict__ bfl,
    const float* __restrict__ bias, float* __restrict__ out) {
    __shared__ float At[16 * 260];
    const int tid  = threadIdx.x;
    const int lane = tid & 31;
    const int wv   = tid >> 5;
    const int m    = lane & 15;
    const int hh   = lane >> 4;
    const int ncol = wv * 16 + m;
    const float bv = bias[ncol];
    const int base_n = blockIdx.x * 16;
    const float4* nf4 = (const float4*)nfeats;
    const float4* ms4 = (const float4*)msg_sum;

    #pragma unroll
    for (int s = 0; s < 4; ++s) {
        int j    = tid + s * 256;
        int row  = j >> 6;
        int c4   = j & 63;
        int node = base_n + row;
        const float4* gp = (c4 < 32)
            ? &nf4[(size_t)node * 32 + c4]
            : &ms4[(size_t)node * 32 + (c4 - 32)];
        async_copy_b128(&At[row * 260 + c4 * 4], gp);
    }
    // per-row mean scale for the h_neigh half
    float d   = deg[base_n + m];
    float inv = (d > 0.0f) ? (1.0f / d) : 0.0f;
    wait_async0();
    __syncthreads();

    v8f acc;
    #pragma unroll
    for (int q = 0; q < 8; ++q) acc[q] = bv;

    #pragma unroll
    for (int kt = 0; kt < 8; ++kt) {
        const float sc = (kt >= 4) ? inv : 1.0f;
        const float* ap = &At[m * 260 + kt * 32 + hh * 16];
        float4 f0 = *(const float4*)(ap + 0);
        float4 f1 = *(const float4*)(ap + 4);
        float4 f2 = *(const float4*)(ap + 8);
        float4 f3 = *(const float4*)(ap + 12);
        v16bf ahv, alv;
        CVT4S(f0, 0, sc) CVT4S(f1, 4, sc) CVT4S(f2, 8, sc) CVT4S(f3, 12, sc)
        const int fi = (kt * 8 + wv) * 32 + lane;
        v16bf bh = bfh[fi];
        v16bf bl = bfl[fi];
        acc = WMMA_BF16(ahv, bh, acc);
        acc = WMMA_BF16(alv, bh, acc);
        acc = WMMA_BF16(ahv, bl, acc);
    }

    const int roff = hh * 8;
    #pragma unroll
    for (int q = 0; q < 8; ++q) {
        int node = base_n + q + roff;
        out[(size_t)node * DD + ncol] = fmaxf(acc[q], 0.0f);
    }
}

extern "C" void kernel_launch(void* const* d_in, const int* in_sizes, int n_in,
                              void* d_out, int out_size, void* d_ws, size_t ws_size,
                              hipStream_t stream) {
    const float* nfeats = (const float*)d_in[0];
    const float* efeats = (const float*)d_in[1];
    const int*   src    = (const int*)d_in[2];
    const int*   dst    = (const int*)d_in[3];
    const float* Wmsg   = (const float*)d_in[4];
    const float* bmsg   = (const float*)d_in[5];
    const float* Wapp   = (const float*)d_in[6];
    const float* bapp   = (const float*)d_in[7];
    float* out = (float*)d_out;

    char* ws = (char*)d_ws;
    float*  msg_sum = (float*)(ws);
    float*  deg     = (float*)(ws + 25600000);
    __bf16* mh      = (__bf16*)(ws + 25800000);
    __bf16* ml      = (__bf16*)(ws + 25865536);
    __bf16* ah      = (__bf16*)(ws + 25931072);
    __bf16* al      = (__bf16*)(ws + 25996608);

    // 1) zero accumulators (msg_sum and deg are contiguous: NN*DD + NN floats)
    sage_zero<<<(NN * DD + NN + 255) / 256, 256, 0, stream>>>(msg_sum, NN * DD + NN);
    // 2) pack weights into WMMA bf16 hi/lo fragments
    sage_wfrag<<<256, 256, 0, stream>>>(Wmsg, Wapp, mh, ml, ah, al);
    // 3) in-degrees
    sage_deg<<<(NE + 255) / 256, 256, 0, stream>>>(dst, deg);
    // 4) edge message GEMM + fused segment-sum scatter  (800k edges / 256 per block)
    sage_edge<<<NE / 256, 256, 0, stream>>>(nfeats, efeats, src, dst,
                                            (const v16bf*)mh, (const v16bf*)ml,
                                            bmsg, msg_sum);
    // 5) mean + apply GEMM + relu  (50k nodes / 16 per block)
    sage_apply<<<NN / 16, 256, 0, stream>>>(nfeats, msg_sum, deg,
                                            (const v16bf*)ah, (const v16bf*)al,
                                            bapp, out);
}
